// VectorQuantizer_30227979829420
// MI455X (gfx1250) — matrix operand order repaired and checked by reading, not compiled, not run on hardware
//
#include <hip/hip_runtime.h>
#include <hip/hip_bf16.h>
#include <math.h>

// ---------------------------------------------------------------------------
// Vector quantizer for MI455X (gfx1250, wave32, WMMA).
// dist argmin via e2 - 2*z.e ; z.e computed with split-bf16 (hi/lo) WMMA:
//   z.e ~= zh.eh + zh.el + zl.eh   (3x v_wmma_f32_16x16x32_bf16)
// ---------------------------------------------------------------------------

typedef __attribute__((ext_vector_type(16))) __bf16 v16bf;
typedef __attribute__((ext_vector_type(8)))  __bf16 v8bf;
typedef __attribute__((ext_vector_type(8)))  float  v8f;

#define B_ROWS  32768
#define K_CODES 8192
#define D_DIM   512
#define BETA_C  0.25f

#define M_BLK   128          // rows of z per block
#define NSPLIT  4            // K-range splits (grid.y)
#define K_PER_SPLIT (K_CODES / NSPLIT)   // 2048
#define SP      520          // padded LDS row stride (bf16 elems) -> bank spread
#define SMEM_BYTES (2 * M_BLK * SP * 2)  // hi+lo, 2B each = 266240 B (<320KB WGP LDS)

// workspace layout (bytes)
#define OFF_CB_HI   ((size_t)0)
#define OFF_CB_LO   ((size_t)8388608)
#define OFF_E2      ((size_t)16777216)
#define OFF_IDX     ((size_t)16809984)
#define OFF_COUNTS  ((size_t)16941056)
#define OFF_SUMSQ   ((size_t)16973824)
#define OFF_CANDV   ((size_t)16974080)
#define OFF_CANDI   ((size_t)17498368)

union ABu { v16bf v; v8bf h[2]; };

__device__ __forceinline__ v8f wmma_bf16(v16bf a, v16bf b, v8f c) {
  // (neg_a, A, neg_b, B, c_mod, C, reuse_a, reuse_b)
  return __builtin_amdgcn_wmma_f32_16x16x32_bf16(false, a, false, b, (short)0, c,
                                                 false, false);
}

// A fragment (16x32 bf16, M=lane%16): lanes<16 hold K {0..15}, lanes>=16 {16..31}
__device__ __forceinline__ v16bf load_a_frag(const __bf16* As, int row, int kk, int lane) {
  const __bf16* p = As + row * SP + kk + ((lane & 16) ? 16 : 0);
  ABu u;
  u.h[0] = *(const v8bf*)(p);        // 16B aligned ds_load_b128
  u.h[1] = *(const v8bf*)(p + 8);
  return u.v;
}

// B fragment (32x16 bf16, N=lane%16): lanes<16 hold K {0..15}, lanes>=16 {16..31}
__device__ __forceinline__ v16bf load_b_frag(const __bf16* cb, int n, int kk, int lane) {
  return *(const v16bf*)(cb + (size_t)n * D_DIM + kk + ((lane & 16) ? 16 : 0));
}

// ---------------------------------------------------------------------------
// Kernel 0: split codebook into bf16 hi/lo, compute per-code ||e||^2
// ---------------------------------------------------------------------------
__global__ void vq_prep_codebook(const float* __restrict__ cb,
                                 __bf16* __restrict__ hi, __bf16* __restrict__ lo,
                                 float* __restrict__ e2) {
  const int k = blockIdx.x;
  const int t = threadIdx.x;                   // 256 threads
  const float* row = cb + (size_t)k * D_DIM;
  float s = 0.f;
  for (int d = t; d < D_DIM; d += 256) {
    float x  = row[d];
    __bf16 h = (__bf16)x;
    __bf16 l = (__bf16)(x - (float)h);
    hi[(size_t)k * D_DIM + d] = h;
    lo[(size_t)k * D_DIM + d] = l;
    s += x * x;
  }
  #pragma unroll
  for (int m = 16; m; m >>= 1) s += __shfl_xor(s, m, 32);
  __shared__ float ws[8];
  if ((t & 31) == 0) ws[t >> 5] = s;
  __syncthreads();
  if (t == 0) {
    float tot = 0.f;
    #pragma unroll
    for (int w = 0; w < 8; ++w) tot += ws[w];
    e2[k] = tot;
  }
}

// ---------------------------------------------------------------------------
// Kernel 1: fused split-bf16 GEMM + running argmin over a K-range.
// grid (B/M_BLK, NSPLIT), 256 threads = 8 waves as 4(M) x 2(N).
// ---------------------------------------------------------------------------
__global__ void __launch_bounds__(256)
vq_gemm_argmin(const float* __restrict__ z,
               const __bf16* __restrict__ cb_hi, const __bf16* __restrict__ cb_lo,
               const float* __restrict__ e2,
               float* __restrict__ cand_v, int* __restrict__ cand_i) {
  extern __shared__ __align__(32) char smem[];
  __bf16* As_hi = (__bf16*)smem;
  __bf16* As_lo = As_hi + M_BLK * SP;

  const int tid    = threadIdx.x;
  const int lane   = tid & 31;
  const int wave   = tid >> 5;
  const int wave_m = wave & 3;       // 0..3 -> 32-row group
  const int wave_n = wave >> 2;      // 0..1 -> 32-col group inside 64-wide chunk
  const int row0   = blockIdx.x * M_BLK;
  const int split  = blockIdx.y;
  const int nsplit0 = split * K_PER_SPLIT;

  // ---- stage this block's 128 z rows as bf16 hi/lo into LDS (once) ----
  for (int i = tid; i < M_BLK * D_DIM; i += 256) {
    int r = i >> 9, c = i & (D_DIM - 1);
    float x  = z[(size_t)(row0 + r) * D_DIM + c];
    __bf16 h = (__bf16)x;
    As_hi[r * SP + c] = h;
    As_lo[r * SP + c] = (__bf16)(x - (float)h);
  }
  __syncthreads();

  const int rowA0 = wave_m * 32 + (lane & 15);       // m-tile 0 rows
  const int rowA1 = rowA0 + 16;                       // m-tile 1 rows

  float rmin[2][8];
  int   ridx[2][8];
  #pragma unroll
  for (int mt = 0; mt < 2; ++mt)
    #pragma unroll
    for (int v = 0; v < 8; ++v) { rmin[mt][v] = 3.0e38f; ridx[mt][v] = 0; }

  const v8f vzero = {0.f,0.f,0.f,0.f,0.f,0.f,0.f,0.f};

  for (int chunk = 0; chunk < K_PER_SPLIT / 64; ++chunk) {
    const int nbase = nsplit0 + chunk * 64 + wave_n * 32;   // this wave's 32 cols
    v8f acc[2][2];
    #pragma unroll
    for (int mt = 0; mt < 2; ++mt)
      #pragma unroll
      for (int nt = 0; nt < 2; ++nt) acc[mt][nt] = vzero;

    for (int kk = 0; kk < D_DIM; kk += 32) {
      v16bf ah[2], al[2], bh[2], bl[2];
      ah[0] = load_a_frag(As_hi, rowA0, kk, lane);
      al[0] = load_a_frag(As_lo, rowA0, kk, lane);
      ah[1] = load_a_frag(As_hi, rowA1, kk, lane);
      al[1] = load_a_frag(As_lo, rowA1, kk, lane);
      const int nb0 = nbase + (lane & 15);
      bh[0] = load_b_frag(cb_hi, nb0,      kk, lane);
      bl[0] = load_b_frag(cb_lo, nb0,      kk, lane);
      bh[1] = load_b_frag(cb_hi, nb0 + 16, kk, lane);
      bl[1] = load_b_frag(cb_lo, nb0 + 16, kk, lane);
      #pragma unroll
      for (int mt = 0; mt < 2; ++mt)
        #pragma unroll
        for (int nt = 0; nt < 2; ++nt) {
          acc[mt][nt] = wmma_bf16(ah[mt], bh[nt], acc[mt][nt]);  // hi*hi
          acc[mt][nt] = wmma_bf16(ah[mt], bl[nt], acc[mt][nt]);  // hi*lo
          acc[mt][nt] = wmma_bf16(al[mt], bh[nt], acc[mt][nt]);  // lo*hi
        }
    }

    // score = e2[n] - 2 * (z.e); running per-lane min (C layout: N = lane%16,
    // half0 VGPR v -> M=v, half1 -> M=v+8)
    #pragma unroll
    for (int nt = 0; nt < 2; ++nt) {
      const int n = nbase + nt * 16 + (lane & 15);
      const float e2n = e2[n];
      #pragma unroll
      for (int mt = 0; mt < 2; ++mt)
        #pragma unroll
        for (int v = 0; v < 8; ++v) {
          float s = fmaf(-2.0f, acc[mt][nt][v], e2n);
          if (s < rmin[mt][v]) { rmin[mt][v] = s; ridx[mt][v] = n; }
        }
    }
  }

  // ---- cross-lane reduce over the 16 N-lanes of each half ----
  __syncthreads();                       // done reading As; reuse LDS
  float* red_v = (float*)smem;           // [M_BLK][2]
  int*   red_i = (int*)(smem + M_BLK * 2 * sizeof(float));
  #pragma unroll
  for (int mt = 0; mt < 2; ++mt)
    #pragma unroll
    for (int v = 0; v < 8; ++v) {
      float val = rmin[mt][v];
      int   idx = ridx[mt][v];
      #pragma unroll
      for (int mask = 8; mask; mask >>= 1) {
        float ov = __shfl_xor(val, mask, 32);
        int   oi = __shfl_xor(idx, mask, 32);
        if (ov < val || (ov == val && oi < idx)) { val = ov; idx = oi; }
      }
      int row = wave_m * 32 + mt * 16 + v + ((lane & 16) ? 8 : 0);
      if ((lane & 15) == 0) {
        red_v[row * 2 + wave_n] = val;
        red_i[row * 2 + wave_n] = idx;
      }
    }
  __syncthreads();

  if (tid < M_BLK) {
    float v0 = red_v[tid * 2 + 0], v1 = red_v[tid * 2 + 1];
    int   i0 = red_i[tid * 2 + 0], i1 = red_i[tid * 2 + 1];
    int   bi = (v1 < v0 || (v1 == v0 && i1 < i0)) ? i1 : i0;
    float bv = (v1 < v0 || (v1 == v0 && i1 < i0)) ? v1 : v0;
    cand_v[(size_t)split * B_ROWS + row0 + tid] = bv;
    cand_i[(size_t)split * B_ROWS + row0 + tid] = bi;
  }
}

// ---------------------------------------------------------------------------
// Kernel 2: merge NSPLIT candidates per row -> final index + histogram
// ---------------------------------------------------------------------------
__global__ void vq_combine(const float* __restrict__ cand_v, const int* __restrict__ cand_i,
                           int* __restrict__ idx_ws, float* __restrict__ out_idx,
                           unsigned int* __restrict__ counts) {
  int b = blockIdx.x * blockDim.x + threadIdx.x;
  if (b >= B_ROWS) return;
  float bv = cand_v[b];
  int   bi = cand_i[b];
  #pragma unroll
  for (int s = 1; s < NSPLIT; ++s) {
    float v = cand_v[(size_t)s * B_ROWS + b];
    int   i = cand_i[(size_t)s * B_ROWS + b];
    if (v < bv || (v == bv && i < bi)) { bv = v; bi = i; }
  }
  idx_ws[b]  = bi;
  out_idx[b] = (float)bi;
  atomicAdd(&counts[bi], 1u);
}

// ---------------------------------------------------------------------------
// Kernel 3: gather quantised vectors (fp32 codebook) + commitment sum
// ---------------------------------------------------------------------------
__global__ void vq_gather_commit(const float* __restrict__ z, const float* __restrict__ cb,
                                 const int* __restrict__ idx, float* __restrict__ out_q,
                                 float* __restrict__ sumsq) {
  const int b = blockIdx.x;
  const int t = threadIdx.x;                 // 128 threads
  const int code = idx[b];
  const float* zr = z + (size_t)b * D_DIM;
  const float* cr = cb + (size_t)code * D_DIM;
  float* qr = out_q + (size_t)b * D_DIM;
  float local = 0.f;
  for (int d = t; d < D_DIM; d += 128) {
    float q = cr[d];
    float diff = zr[d] - q;
    qr[d] = q;                                // quantised_st forward value == q
    local += diff * diff;
  }
  #pragma unroll
  for (int m = 16; m; m >>= 1) local += __shfl_xor(local, m, 32);
  __shared__ float ws4[4];
  if ((t & 31) == 0) ws4[t >> 5] = local;
  __syncthreads();
  if (t == 0) atomicAdd(sumsq, ws4[0] + ws4[1] + ws4[2] + ws4[3]);
}

// ---------------------------------------------------------------------------
// Kernel 4: entropy, codes_used, commit scalar
// ---------------------------------------------------------------------------
__global__ void vq_finalize(const unsigned int* __restrict__ counts,
                            const float* __restrict__ sumsq,
                            float* __restrict__ out_scalars) {
  const int t = threadIdx.x;                 // 256 threads
  float ent = 0.f;
  int used = 0;
  for (int k = t; k < K_CODES; k += 256) {
    unsigned int c = counts[k];
    float u = (float)c * (1.0f / (float)B_ROWS) + 1e-10f;
    ent -= u * logf(u);
    used += (c > 0u) ? 1 : 0;
  }
  #pragma unroll
  for (int m = 16; m; m >>= 1) {
    ent  += __shfl_xor(ent,  m, 32);
    used += __shfl_xor(used, m, 32);
  }
  __shared__ float we[8];
  __shared__ int   wu[8];
  if ((t & 31) == 0) { we[t >> 5] = ent; wu[t >> 5] = used; }
  __syncthreads();
  if (t == 0) {
    float E = 0.f; int U = 0;
    #pragma unroll
    for (int w = 0; w < 8; ++w) { E += we[w]; U += wu[w]; }
    out_scalars[0] = BETA_C * sumsq[0] / (float)((size_t)B_ROWS * D_DIM);  // commit
    out_scalars[1] = E;                                                    // entropy
    out_scalars[2] = (float)U;                                             // codes_used
  }
}

// ---------------------------------------------------------------------------
extern "C" void kernel_launch(void* const* d_in, const int* in_sizes, int n_in,
                              void* d_out, int out_size, void* d_ws, size_t ws_size,
                              hipStream_t stream) {
  (void)in_sizes; (void)n_in; (void)out_size; (void)ws_size;
  const float* z  = (const float*)d_in[0];   // (B, D)
  const float* cb = (const float*)d_in[1];   // (K, D)

  char* ws = (char*)d_ws;
  __bf16*       cb_hi  = (__bf16*)(ws + OFF_CB_HI);
  __bf16*       cb_lo  = (__bf16*)(ws + OFF_CB_LO);
  float*        e2     = (float*)(ws + OFF_E2);
  int*          idx_ws = (int*)(ws + OFF_IDX);
  unsigned int* counts = (unsigned int*)(ws + OFF_COUNTS);
  float*        sumsq  = (float*)(ws + OFF_SUMSQ);
  float*        cand_v = (float*)(ws + OFF_CANDV);
  int*          cand_i = (int*)(ws + OFF_CANDI);

  float* out     = (float*)d_out;
  float* out_q   = out;                                  // (B, D)
  float* out_idx = out + (size_t)B_ROWS * D_DIM;         // (B,)
  float* out_scl = out_idx + B_ROWS;                     // commit, entropy, codes_used

  // zero counts + sumsq (contiguous region)
  hipMemsetAsync(ws + OFF_COUNTS, 0, K_CODES * sizeof(unsigned int) + 256, stream);

  vq_prep_codebook<<<K_CODES, 256, 0, stream>>>(cb, cb_hi, cb_lo, e2);

  hipFuncSetAttribute((const void*)vq_gemm_argmin,
                      hipFuncAttributeMaxDynamicSharedMemorySize, SMEM_BYTES);
  dim3 ggrid(B_ROWS / M_BLK, NSPLIT);
  vq_gemm_argmin<<<ggrid, 256, SMEM_BYTES, stream>>>(z, cb_hi, cb_lo, e2, cand_v, cand_i);

  vq_combine<<<(B_ROWS + 255) / 256, 256, 0, stream>>>(cand_v, cand_i, idx_ws, out_idx, counts);
  vq_gather_commit<<<B_ROWS, 128, 0, stream>>>(z, cb, idx_ws, out_q, sumsq);
  vq_finalize<<<1, 256, 0, stream>>>(counts, sumsq, out_scl);
}
